// WindowAttention_76235669504701
// MI455X (gfx1250) — matrix operand order repaired and checked by reading, not compiled, not run on hardware
//
#include <hip/hip_runtime.h>

typedef _Float16 f16;
typedef _Float16 v16h __attribute__((ext_vector_type(16)));
typedef _Float16 v8h  __attribute__((ext_vector_type(8)));
typedef float    v8f  __attribute__((ext_vector_type(8)));
typedef float    v4f  __attribute__((ext_vector_type(4)));
typedef unsigned int v4u __attribute__((ext_vector_type(4)));
typedef int          v4i __attribute__((ext_vector_type(4)));
typedef int          v8i __attribute__((ext_vector_type(8)));

#define WMMA_F16(a, b, c) \
  __builtin_amdgcn_wmma_f32_16x16x32_f16(false, (a), false, (b), (short)0, (c), false, false)

// ---------- constants ----------
#define NTOK 49
#define HEADS 12
#define HD 32
#define DIM 384
#define MTOT 401408          // 8192 * 49
#define SCALE 0.17677669529663687f
#define QK_STRIDE 1568       // 49*32 halves per (window,head)
#define VT_STRIDE 2048       // 32*64 halves per (window,head), keys padded to 64

// LDS layout for the TDM-staged GEMMs (dynamic LDS, bytes)
#define LDS_B0 0u            // B slab buf0: 384 rows x 32 halves = 24576 B
#define LDS_B1 24576u
#define LDS_A0 49152u        // A slab buf0: 64 rows x 32 halves = 4096 B
#define LDS_A1 53248u
#define LDS_TOTAL 57344u

// ---------- fragment loaders (per CDNA5 ISA 7.12.2 VGPR layouts) ----------

// A fragment 16x32 f16 from row-major f16 src (row stride ld halves).
__device__ __forceinline__ v16h load_a_f16(const f16* base, int ld, int m0, int k0) {
  const int l  = threadIdx.x & 31;
  const int m  = m0 + (l & 15);
  const int kb = k0 + ((l >> 4) << 3);
  const f16* p = base + (size_t)m * ld + kb;
  v8h lo = *(const v8h*)p;
  v8h hi = *(const v8h*)(p + 16);
  v16h a;
#pragma unroll
  for (int i = 0; i < 8; ++i) { a[i] = lo[i]; a[i + 8] = hi[i]; }
  return a;
}

// B fragment 32x16 f16. S is "B-source": S[n][c] = B[c][n], row-major, stride ld.
__device__ __forceinline__ v16h load_b_f16(const f16* S, int ld, int n0, int k0) {
  const int l  = threadIdx.x & 31;
  const int n  = n0 + (l & 15);
  const int kb = k0 + ((l >> 4) << 3);
  const f16* p = S + (size_t)n * ld + kb;
  v8h lo = *(const v8h*)p;
  v8h hi = *(const v8h*)(p + 16);
  v16h b;
#pragma unroll
  for (int i = 0; i < 8; ++i) { b[i] = lo[i]; b[i + 8] = hi[i]; }
  return b;
}

// ---------- TDM: async DMA of a 2D f16 tile (tile_k x tile_rows) into LDS ----------
// Descriptor per CDNA5 ISA (count=1, type=2, 2B elems).
__device__ __forceinline__ void tdm_load_2d(unsigned lds_byte, const f16* gptr,
                                            int tile_k, int tile_rows,
                                            int tensor_w, int tensor_h) {
  unsigned long long ga = (unsigned long long)(uintptr_t)gptr;
  v4u g0;
  g0[0] = 1u;                                            // count=1, user descriptor
  g0[1] = lds_byte;                                      // lds_addr
  g0[2] = (unsigned)(ga & 0xFFFFFFFFu);                  // global_addr[31:0]
  g0[3] = (unsigned)((ga >> 32) & 0x01FFFFFFu) | (2u << 30); // addr[56:32] | type=2
  v8i g1;
  g1[0] = (1 << 16);                                     // data_size = 1 (2 bytes)
  g1[1] = (tensor_w & 0xFFFF) << 16;                     // tensor_dim0[15:0]
  g1[2] = ((tensor_w >> 16) & 0xFFFF) | ((tensor_h & 0xFFFF) << 16);
  g1[3] = ((tensor_h >> 16) & 0xFFFF) | (tile_k << 16);  // tensor_dim1 hi | tile_dim0
  g1[4] = tile_rows & 0xFFFF;                            // tile_dim1 (tile_dim2 = 0)
  g1[5] = tensor_w;                                      // tensor_dim0_stride lo32
  g1[6] = 0;
  g1[7] = 0;
  v4i g2 = {0, 0, 0, 0};
  v4i g3 = {0, 0, 0, 0};
#if defined(__clang_major__) && __clang_major__ >= 23
  v8i g4 = {0, 0, 0, 0, 0, 0, 0, 0};
  __builtin_amdgcn_tensor_load_to_lds(g0, g1, g2, g3, g4, 0);
#else
  __builtin_amdgcn_tensor_load_to_lds(g0, g1, g2, g3, 0);
#endif
}

// ---------- shared GEMM mainloop: C(64x384 per block) += A(64x384) * W^T ----------
// 8 waves: 4 M-tiles x 2 N-halves; per-wave 16 rows x 192 cols (12 accum frags).
// TDM double-buffers a 384x32 B slab + 64x32 A slab per K-step.
// B-fragment ds_loads are software-pipelined two deep to overlap DS latency.
__device__ __forceinline__ void gemm_mainloop(const f16* Asrc, const f16* Wsrc,
                                              int m0_blk, int nbase_blk, int w_rows,
                                              f16* lds, v8f acc[12]) {
  const int wave = threadIdx.x >> 5;
  const unsigned ldsbase = (unsigned)(uintptr_t)lds;  // LDS byte offset (flat addr[31:0])

  if (wave == 0) {
    tdm_load_2d(ldsbase + LDS_B0, Wsrc + (size_t)nbase_blk * DIM, 32, 384, DIM, w_rows);
    tdm_load_2d(ldsbase + LDS_A0, Asrc + (size_t)m0_blk * DIM, 32, 64, DIM, MTOT);
  }
  int buf = 0;
  for (int ks = 0; ks < 12; ++ks) {
    if (wave == 0) __builtin_amdgcn_s_wait_tensorcnt(0);
    __syncthreads();                       // current buffers ready for all waves
    if (wave == 0 && ks + 1 < 12) {        // kick off next K-slab into other buffer
      const int k0 = (ks + 1) * 32;
      tdm_load_2d(ldsbase + (buf ? LDS_B0 : LDS_B1),
                  Wsrc + (size_t)nbase_blk * DIM + k0, 32, 384, DIM, w_rows);
      tdm_load_2d(ldsbase + (buf ? LDS_A0 : LDS_A1),
                  Asrc + (size_t)m0_blk * DIM + k0, 32, 64, DIM, MTOT);
    }
    const f16* Abuf = lds + (buf ? LDS_A1 : LDS_A0) / 2;
    const f16* Bbuf = lds + (buf ? LDS_B1 : LDS_B0) / 2;
    v16h a = load_a_f16(Abuf, 32, (wave & 3) * 16, 0);
    const int nw = (wave >> 2) * 192;
    // 2-deep rotating B-fragment pipeline
    v16h b0 = load_b_f16(Bbuf, 32, nw, 0);
    v16h b1 = load_b_f16(Bbuf, 32, nw + 16, 0);
#pragma unroll
    for (int nt = 0; nt < 12; ++nt) {
      v16h bn = (nt + 2 < 12) ? load_b_f16(Bbuf, 32, nw + (nt + 2) * 16, 0) : b0;
      acc[nt] = WMMA_F16(a, b0, acc[nt]);
      b0 = b1;
      b1 = bn;
    }
    __syncthreads();                       // all reads done before buffer is reused
    buf ^= 1;
  }
}

// ---------- kernel 0: f32 -> f16 pre-conversion (8 elems / thread) ----------
__global__ __launch_bounds__(256) void cvt_f32_f16(const float* __restrict__ src,
                                                   f16* __restrict__ dst, int n8) {
  const int i = blockIdx.x * 256 + threadIdx.x;
  if (i >= n8) return;
  const v4f* p = (const v4f*)(src + (size_t)i * 8);
  v4f a = p[0], b = p[1];
  v8h o;
#pragma unroll
  for (int j = 0; j < 4; ++j) { o[j] = (f16)a[j]; o[j + 4] = (f16)b[j]; }
  *(v8h*)(dst + (size_t)i * 8) = o;
}

// ---------- kernel 1: qkv = x @ qkv_w^T + b -> q(scaled), k, v^T (f16) ----------
// grid = (MTOT/64, 1152/384)
__global__ __launch_bounds__(256) void qkv_gemm(const f16* __restrict__ xh,
                                                const f16* __restrict__ wh,
                                                const float* __restrict__ bias,
                                                f16* __restrict__ q,
                                                f16* __restrict__ kmat,
                                                f16* __restrict__ vT) {
  extern __shared__ f16 ldsmem[];
  const int wave  = threadIdx.x >> 5;
  const int lane  = threadIdx.x & 31;
  const int m0    = blockIdx.x * 64 + (wave & 3) * 16;
  const int nbase = blockIdx.y * 384 + (wave >> 2) * 192;

  v8f acc[12] = {};
  gemm_mainloop(xh, wh, blockIdx.x * 64, blockIdx.y * 384, 3 * DIM, ldsmem, acc);

  const int nloc = lane & 15;
  const int hi8  = (lane >> 4) << 3;
#pragma unroll
  for (int nt = 0; nt < 12; ++nt) {
    const int j     = nbase + nt * 16 + nloc;
    const int three = j / DIM;
    const int hcol  = j - three * DIM;
    const int h     = hcol >> 5;
    const int hd    = hcol & 31;
    const float bj  = bias[j];
#pragma unroll
    for (int e = 0; e < 8; ++e) {
      const int t    = m0 + e + hi8;
      const int bwin = t / NTOK;
      const int nn   = t - bwin * NTOK;
      const size_t bh = (size_t)bwin * HEADS + h;
      const float val = acc[nt][e] + bj;
      if (three == 0) {
        q[bh * QK_STRIDE + nn * HD + hd] = (f16)(val * SCALE);
      } else if (three == 1) {
        kmat[bh * QK_STRIDE + nn * HD + hd] = (f16)val;
      } else {
        vT[bh * VT_STRIDE + hd * 64 + nn] = (f16)val;
      }
    }
  }
}

// ---------- kernel 2: fused attention per (window, head) ----------
__global__ __launch_bounds__(128) void win_attn(const f16* __restrict__ q,
                                                const f16* __restrict__ kmat,
                                                const f16* __restrict__ vT,
                                                const float* __restrict__ mask,
                                                const float* __restrict__ table,
                                                const int* __restrict__ rindex,
                                                f16* __restrict__ ao) {
  __shared__ f16 smem[4 * 16 * 64];
  const int wid  = threadIdx.x >> 5;
  const int lane = threadIdx.x & 31;
  const int bh   = blockIdx.x;
  const int bwin = bh / HEADS;
  const int h    = bh - bwin * HEADS;
  const int wmsk = bwin & 63;
  const int m0   = wid * 16;

  const f16* qb = q    + (size_t)bh * QK_STRIDE;
  const f16* kb = kmat + (size_t)bh * QK_STRIDE;
  const f16* vb = vT   + (size_t)bh * VT_STRIDE;

  // ---- attn = (q*scale) @ k^T, one 16x64 slab per wave, K = HD = 32
  v16h aq = load_a_f16(qb, HD, m0, 0);
  v8f att[4];
#pragma unroll
  for (int nt = 0; nt < 4; ++nt) {
    v16h bkf = load_b_f16(kb, HD, nt * 16, 0);
    v8f z = {};
    att[nt] = WMMA_F16(aq, bkf, z);
  }

  const int nloc = lane & 15;
  const int hi8  = (lane >> 4) << 3;

  // ---- + rel-pos bias + shift mask; pad cols >= 49 to -1e30
#pragma unroll
  for (int nt = 0; nt < 4; ++nt) {
    const int n = nt * 16 + nloc;
#pragma unroll
    for (int e = 0; e < 8; ++e) {
      const int m = m0 + e + hi8;
      float val = att[nt][e];
      if (n >= NTOK) {
        val = -1e30f;
      } else if (m < NTOK) {
        const int idx = rindex[m * NTOK + n];
        val += table[idx * HEADS + h] + mask[(size_t)wmsk * (NTOK * NTOK) + m * NTOK + n];
      }
      att[nt][e] = val;
    }
  }

  // ---- row softmax over 64 cols: 4 per-lane values + xor-reduce over 16-lane half
#pragma unroll
  for (int e = 0; e < 8; ++e) {
    float mx = fmaxf(fmaxf(att[0][e], att[1][e]), fmaxf(att[2][e], att[3][e]));
    mx = fmaxf(mx, __shfl_xor(mx, 1));
    mx = fmaxf(mx, __shfl_xor(mx, 2));
    mx = fmaxf(mx, __shfl_xor(mx, 4));
    mx = fmaxf(mx, __shfl_xor(mx, 8));
    float s = 0.f;
#pragma unroll
    for (int nt = 0; nt < 4; ++nt) {
      const float ev = __expf(att[nt][e] - mx);
      att[nt][e] = ev;
      s += ev;
    }
    s += __shfl_xor(s, 1);
    s += __shfl_xor(s, 2);
    s += __shfl_xor(s, 4);
    s += __shfl_xor(s, 8);
    const float inv = 1.f / s;
#pragma unroll
    for (int nt = 0; nt < 4; ++nt) att[nt][e] *= inv;
  }

  // ---- restage C-layout -> A-layout via per-wave LDS tile (16x64 f16)
  f16* tile = smem + wid * (16 * 64);
#pragma unroll
  for (int nt = 0; nt < 4; ++nt) {
    const int n = nt * 16 + nloc;
#pragma unroll
    for (int e = 0; e < 8; ++e) {
      tile[(e + hi8) * 64 + n] = (f16)att[nt][e];
    }
  }
  // same-wave DS ops are in-order; VGPR deps force the dscnt wait

  // ---- out = attn @ v : A = 16x64 probs (2 K-steps), B = v^T rows (contiguous)
  v16h a0 = load_a_f16(tile, 64, 0, 0);
  v16h a1 = load_a_f16(tile, 64, 0, 32);
  v8f outf[2];
#pragma unroll
  for (int nt = 0; nt < 2; ++nt) {
    v16h b0 = load_b_f16(vb, 64, nt * 16, 0);
    v16h b1 = load_b_f16(vb, 64, nt * 16, 32);
    v8f z = {};
    z = WMMA_F16(a0, b0, z);
    outf[nt] = WMMA_F16(a1, b1, z);
  }

  // ---- store to (token, 384) f16, proj-ready layout
#pragma unroll
  for (int nt = 0; nt < 2; ++nt) {
#pragma unroll
    for (int e = 0; e < 8; ++e) {
      const int m = m0 + e + hi8;
      if (m < NTOK) {
        ao[((size_t)bwin * NTOK + m) * DIM + h * HD + nt * 16 + nloc] = (f16)outf[nt][e];
      }
    }
  }
}

// ---------- kernel 3: out = ao @ proj_w^T + proj_b (f32 out), grid = (MTOT/64, 1) ----------
__global__ __launch_bounds__(256) void proj_gemm(const f16* __restrict__ ao,
                                                 const f16* __restrict__ wh,
                                                 const float* __restrict__ bias,
                                                 float* __restrict__ out) {
  extern __shared__ f16 ldsmem[];
  const int wave  = threadIdx.x >> 5;
  const int lane  = threadIdx.x & 31;
  const int m0    = blockIdx.x * 64 + (wave & 3) * 16;
  const int nbase = (wave >> 2) * 192;

  v8f acc[12] = {};
  gemm_mainloop(ao, wh, blockIdx.x * 64, 0, DIM, ldsmem, acc);

  const int nloc = lane & 15;
  const int hi8  = (lane >> 4) << 3;
#pragma unroll
  for (int nt = 0; nt < 12; ++nt) {
    const int j    = nbase + nt * 16 + nloc;
    const float bj = bias[j];
#pragma unroll
    for (int e = 0; e < 8; ++e) {
      const int t = m0 + e + hi8;
      out[(size_t)t * DIM + j] = acc[nt][e] + bj;
    }
  }
}

extern "C" void kernel_launch(void* const* d_in, const int* in_sizes, int n_in,
                              void* d_out, int out_size, void* d_ws, size_t ws_size,
                              hipStream_t stream) {
  (void)in_sizes; (void)n_in; (void)out_size; (void)ws_size;
  const float* x      = (const float*)d_in[0];
  const float* mask   = (const float*)d_in[1];
  const float* table  = (const float*)d_in[2];
  const float* qkv_w  = (const float*)d_in[3];
  const float* qkv_b  = (const float*)d_in[4];
  const float* proj_w = (const float*)d_in[5];
  const float* proj_b = (const float*)d_in[6];
  const int*   rindex = (const int*)d_in[7];
  float* out = (float*)d_out;

  const size_t BH   = 98304;                 // 8192 windows * 12 heads
  const size_t XSZ  = (size_t)MTOT * DIM;    // 154,140,672
  const size_t WQSZ = (size_t)3 * DIM * DIM; // 442,368
  const size_t WPSZ = (size_t)DIM * DIM;     // 147,456

  f16* xh   = (f16*)d_ws;                 // x in f16
  f16* wqh  = xh   + XSZ;                 // qkv_w in f16
  f16* wph  = wqh  + WQSZ;                // proj_w in f16
  f16* q    = wph  + WPSZ;                // BH * 1568
  f16* kmat = q    + BH * QK_STRIDE;      // BH * 1568
  f16* vT   = kmat + BH * QK_STRIDE;      // BH * 2048 (keys padded to 64)
  f16* ao   = vT   + BH * VT_STRIDE;      // MTOT * 384
  // total workspace ~1.55 GB

  // zero v^T so padded key columns (49..63) are exactly 0 in the PV matmul
  hipMemsetAsync(vT, 0, BH * VT_STRIDE * sizeof(f16), stream);

  cvt_f32_f16<<<(unsigned)((XSZ / 8 + 255) / 256), 256, 0, stream>>>(x, xh, (int)(XSZ / 8));
  cvt_f32_f16<<<(unsigned)((WQSZ / 8 + 255) / 256), 256, 0, stream>>>(qkv_w, wqh, (int)(WQSZ / 8));
  cvt_f32_f16<<<(unsigned)((WPSZ / 8 + 255) / 256), 256, 0, stream>>>(proj_w, wph, (int)(WPSZ / 8));

  qkv_gemm<<<dim3(MTOT / 64, 3), 256, LDS_TOTAL, stream>>>(xh, wqh, qkv_b, q, kmat, vT);
  win_attn<<<dim3((unsigned)BH), 128, 0, stream>>>(q, kmat, vT, mask, table, rindex, ao);
  proj_gemm<<<dim3(MTOT / 64, 1), 256, LDS_TOTAL, stream>>>(ao, wph, proj_b, out);
}
